// MLAPlusMoEBlock_65412351918132
// MI455X (gfx1250) — matrix operand order repaired and checked by reading, not compiled, not run on hardware
//
#include <hip/hip_runtime.h>
#include <hip/hip_bf16.h>

// ---------------------------------------------------------------------------
// MLA + MoE block for gfx1250 (MI455X). All heavy GEMMs use
// v_wmma_f32_16x16x32_f16 (wave32 WMMA, fp32 accumulate). GEMM A-tiles are
// staged with CDNA5 async global->LDS copies (global_load_async_to_lds_b128,
// ASYNCcnt), double-buffered to overlap data movement with WMMA issue.
// ---------------------------------------------------------------------------

typedef __attribute__((ext_vector_type(16))) _Float16 v16h;
typedef __attribute__((ext_vector_type(8)))  float    v8f;

#define S_LEN 2048
#define D_DIM 2048
#define H_N   16
#define QR_N  1536
#define KVR_N 512
#define DN_N  128
#define DR_N  64
#define DV_N  128
#define E_N   8
#define F_N   1408
#define DK_N  (DN_N + DR_N)        // 192
#define QB_COLS  (H_N * (DN_N + DR_N))   // 3072
#define KVA_COLS (KVR_N + DR_N)          // 576
#define KVB_COLS (H_N * (DN_N + DV_N))   // 4096
#define HD_COLS  (H_N * DV_N)            // 2048

// ---------------------------------------------------------------------------
// CDNA5 async global -> LDS copy (16B per lane), tracked by ASYNCcnt.
// ---------------------------------------------------------------------------
__device__ __forceinline__ void async_copy16(unsigned lds_addr,
                                             const void* gaddr) {
  asm volatile("global_load_async_to_lds_b128 %0, %1, off"
               :: "v"(lds_addr), "v"(gaddr)
               : "memory");
}
__device__ __forceinline__ void wait_async0() {
  asm volatile("s_wait_asynccnt 0x0" ::: "memory");
}

// ---------------------------------------------------------------------------
// Elementwise helpers
// ---------------------------------------------------------------------------
__global__ void f32_to_f16_kernel(const float* __restrict__ in,
                                  _Float16* __restrict__ out, size_t n) {
  size_t i = (size_t)blockIdx.x * blockDim.x + threadIdx.x;
  size_t stride = (size_t)gridDim.x * blockDim.x;
  for (; i < n; i += stride) out[i] = (_Float16)in[i];
}

__global__ void add_f32_kernel(const float* __restrict__ a,
                               const float* __restrict__ b,
                               float* __restrict__ c, size_t n) {
  size_t i = (size_t)blockIdx.x * blockDim.x + threadIdx.x;
  size_t stride = (size_t)gridDim.x * blockDim.x;
  for (; i < n; i += stride) c[i] = a[i] + b[i];
}

__global__ void silu_mul_kernel(const float* __restrict__ g,
                                const float* __restrict__ u,
                                _Float16* __restrict__ a, size_t n) {
  size_t i = (size_t)blockIdx.x * blockDim.x + threadIdx.x;
  size_t stride = (size_t)gridDim.x * blockDim.x;
  for (; i < n; i += stride) {
    float gv = g[i];
    float sv = gv / (1.0f + __expf(-gv));
    a[i] = (_Float16)(sv * u[i]);
  }
}

__global__ void moe_accum_kernel(const float* __restrict__ eo,
                                 const float* __restrict__ combine,
                                 float* __restrict__ mo, int e, size_t n) {
  size_t i = (size_t)blockIdx.x * blockDim.x + threadIdx.x;
  size_t stride = (size_t)gridDim.x * blockDim.x;
  for (; i < n; i += stride) {
    size_t t = i / D_DIM;
    mo[i] += combine[t * E_N + e] * eo[i];
  }
}

// ---------------------------------------------------------------------------
// RMSNorm: fp32 row input (strided) -> f16 row output
// ---------------------------------------------------------------------------
__global__ __launch_bounds__(256)
void rmsnorm_f16_kernel(const float* __restrict__ in, int in_stride,
                        const float* __restrict__ w,
                        _Float16* __restrict__ out, int out_stride, int dim) {
  int row = blockIdx.x;
  const float* x = in + (size_t)row * in_stride;
  float ss = 0.0f;
  for (int i = threadIdx.x; i < dim; i += 256) { float v = x[i]; ss += v * v; }
  for (int m = 16; m >= 1; m >>= 1) ss += __shfl_xor(ss, m, 32);
  __shared__ float red[8];
  if ((threadIdx.x & 31) == 0) red[threadIdx.x >> 5] = ss;
  __syncthreads();
  if (threadIdx.x == 0) {
    float t = 0.0f;
    for (int i = 0; i < 8; ++i) t += red[i];
    red[0] = t;
  }
  __syncthreads();
  float rstd = rsqrtf(red[0] / (float)dim + 1e-5f);
  for (int i = threadIdx.x; i < dim; i += 256)
    out[(size_t)row * out_stride + i] = (_Float16)(x[i] * rstd * w[i]);
}

// ---------------------------------------------------------------------------
// Guarded 8-half load of B row segment (zero-fill past N).
// ---------------------------------------------------------------------------
__device__ __forceinline__ uint4 loadB_guard(const _Float16* __restrict__ B,
                                             int krow, int gn, int N) {
  if (gn + 7 < N)
    return *reinterpret_cast<const uint4*>(B + (size_t)krow * N + gn);
  uint4 r;
  _Float16* t = (_Float16*)&r;
  for (int i = 0; i < 8; ++i)
    t[i] = (gn + i < N) ? B[(size_t)krow * N + gn + i] : (_Float16)0.0f;
  return r;
}

// ---------------------------------------------------------------------------
// Generic f16 GEMM: C[M][N] (f32) = A[M][K] * B[K][N], tiles 128x128x32.
// 256 threads = 8 waves; each wave: 32x64 subtile = 2x4 WMMA accumulators.
// Double-buffered LDS. A tile staged via async global->LDS (no VGPR
// round-trip); B tile prefetched to registers, then transposed into LDS
// [n][k] so both A and B fragment loads are contiguous ds_load_b128 matching
// one shared contiguous k order (k = 16*lg + s) for A and B.
// ---------------------------------------------------------------------------
__global__ __launch_bounds__(256)
void gemm_f16_kernel(const _Float16* __restrict__ A,
                     const _Float16* __restrict__ B,
                     float* __restrict__ C, int M, int N, int K) {
  __shared__ __align__(16) _Float16 As[2][128][40];  // [m][k], pad to 40
  __shared__ __align__(16) _Float16 Bs[2][128][40];  // [n][k]
  const int tid  = threadIdx.x;
  const int lane = tid & 31;
  const int wid  = tid >> 5;
  const int lg   = lane >> 4;   // 0/1 lane group
  const int ll   = lane & 15;
  const int bm = blockIdx.y * 128;
  const int bn = blockIdx.x * 128;
  const int wm = (wid & 3) * 32;   // wave row offset in tile
  const int wn = (wid >> 2) * 64;  // wave col offset in tile

  // staging assignments: each thread moves 2x16B of A and 2x16B of B per tile
  const int ar0 = tid >> 2,        ac0 = (tid & 3) * 8;   // rows 0..63
  const int ar1 = 64 + (tid >> 2), ac1 = ac0;             // rows 64..127
  const int bk0 = tid >> 4,        bn0 = (tid & 15) * 8;  // k 0..15
  const int bk1 = 16 + (tid >> 4), bn1 = bn0;             // k 16..31

  const int nk = K >> 5;

  v8f acc[2][4];
  for (int mf = 0; mf < 2; ++mf)
    for (int nf = 0; nf < 4; ++nf)
      for (int i = 0; i < 8; ++i) acc[mf][nf][i] = 0.0f;

  // ---- prologue: stage tile 0 ----
  async_copy16((unsigned)(size_t)&As[0][ar0][ac0],
               A + (size_t)(bm + ar0) * K + ac0);
  async_copy16((unsigned)(size_t)&As[0][ar1][ac1],
               A + (size_t)(bm + ar1) * K + ac1);
  uint4 bv0 = loadB_guard(B, bk0, bn + bn0, N);
  uint4 bv1 = loadB_guard(B, bk1, bn + bn1, N);

  for (int kt = 0; kt < nk; ++kt) {
    const int buf = kt & 1;
    // complete this tile's async A copies (own-wave), commit B to LDS
    wait_async0();
    {
      const _Float16* t0 = (const _Float16*)&bv0;
      const _Float16* t1 = (const _Float16*)&bv1;
      for (int i = 0; i < 8; ++i) Bs[buf][bn0 + i][bk0] = t0[i];
      for (int i = 0; i < 8; ++i) Bs[buf][bn1 + i][bk1] = t1[i];
    }
    __syncthreads();

    // issue next tile into the other buffer while we compute this one
    if (kt + 1 < nk) {
      const int k0n = (kt + 1) * 32;
      async_copy16((unsigned)(size_t)&As[buf ^ 1][ar0][ac0],
                   A + (size_t)(bm + ar0) * K + k0n + ac0);
      async_copy16((unsigned)(size_t)&As[buf ^ 1][ar1][ac1],
                   A + (size_t)(bm + ar1) * K + k0n + ac1);
      bv0 = loadB_guard(B, k0n + bk0, bn + bn0, N);
      bv1 = loadB_guard(B, k0n + bk1, bn + bn1, N);
    }

    // ---- fragments + WMMA ----
    v16h afr[2], bfr[4];
    for (int mf = 0; mf < 2; ++mf) {
      int row = wm + mf * 16 + ll;
      uint4* d = reinterpret_cast<uint4*>(&afr[mf]);
      d[0] = *reinterpret_cast<const uint4*>(&As[buf][row][16 * lg]);
      d[1] = *reinterpret_cast<const uint4*>(&As[buf][row][8 + 16 * lg]);
    }
    for (int nf = 0; nf < 4; ++nf) {
      int col = wn + nf * 16 + ll;
      uint4* d = reinterpret_cast<uint4*>(&bfr[nf]);
      const uint4* s = reinterpret_cast<const uint4*>(&Bs[buf][col][16 * lg]);
      d[0] = s[0];
      d[1] = s[1];
    }
    for (int mf = 0; mf < 2; ++mf)
      for (int nf = 0; nf < 4; ++nf)
        acc[mf][nf] = __builtin_amdgcn_wmma_f32_16x16x32_f16(
            false, afr[mf], false, bfr[nf], (short)0, acc[mf][nf], false, false);
    __syncthreads();
  }

  // ---- store C (C-layout: lane ll = N col, VGPR v + 8*lg = M row) ----
  for (int mf = 0; mf < 2; ++mf) {
    int rbase = bm + wm + mf * 16 + lg * 8;
    for (int nf = 0; nf < 4; ++nf) {
      int col = bn + wn + nf * 16 + ll;
      if (col < N) {
        for (int v = 0; v < 8; ++v)
          C[(size_t)(rbase + v) * N + col] = acc[mf][nf][v];
      }
    }
  }
}

// ---------------------------------------------------------------------------
// RoPE + layout prep: builds qf [S][H][192] f16 (scaled by 1/sqrt(192)),
// kf [S][H][192] f16, and vT [H][128][S] f16 (transposed for PV WMMA frags).
// ---------------------------------------------------------------------------
__global__ __launch_bounds__(256)
void prep_qkv_kernel(const float* __restrict__ q, const float* __restrict__ kv,
                     const float* __restrict__ ckv,
                     _Float16* __restrict__ qf, _Float16* __restrict__ kf,
                     _Float16* __restrict__ vT) {
  int t = blockIdx.x;
  const float scale = rsqrtf((float)DK_N);
  const float LOG1E4 = 9.210340371976184f;  // ln(10000)
  for (int i = threadIdx.x; i < H_N * DK_N; i += 256) {
    int hh = i / DK_N, d = i % DK_N;
    float qo, ko;
    if (d < DN_N) {
      qo = q[(size_t)t * QB_COLS + hh * DK_N + d];
      ko = kv[(size_t)t * KVB_COLS + hh * (DN_N + DV_N) + d];
    } else {
      int j = d - DN_N;               // 0..63
      int p = (j < 32) ? j : j - 32;  // pair index
      float inv = __expf(-((float)p / 32.0f) * LOG1E4);
      float ang = (float)t * inv;
      float cv = cosf(ang), sv = sinf(ang);
      const float* qp = q + (size_t)t * QB_COLS + hh * DK_N + DN_N;
      const float* kp = ckv + (size_t)t * KVA_COLS + KVR_N;
      if (j < 32) {
        qo = qp[2 * p] * cv - qp[2 * p + 1] * sv;
        ko = kp[2 * p] * cv - kp[2 * p + 1] * sv;
      } else {
        qo = qp[2 * p + 1] * cv + qp[2 * p] * sv;
        ko = kp[2 * p + 1] * cv + kp[2 * p] * sv;
      }
    }
    qf[((size_t)t * H_N + hh) * DK_N + d] = (_Float16)(qo * scale);
    kf[((size_t)t * H_N + hh) * DK_N + d] = (_Float16)ko;
  }
  for (int i = threadIdx.x; i < H_N * DV_N; i += 256) {
    int hh = i >> 7, dv = i & 127;
    float v = kv[(size_t)t * KVB_COLS + hh * (DN_N + DV_N) + DN_N + dv];
    vT[((size_t)hh * DV_N + dv) * S_LEN + t] = (_Float16)v;
  }
}

// ---------------------------------------------------------------------------
// Flash attention per (head, 128-query block). 8 waves; each wave owns 16 q
// rows x full 128 dv. Key blocks of 64. Q frags in registers, K/V frags
// loaded directly from global; P round-trips through per-wave LDS tile.
// ---------------------------------------------------------------------------
__global__ __launch_bounds__(256)
void mla_attn_kernel(const _Float16* __restrict__ qf,
                     const _Float16* __restrict__ kf,
                     const _Float16* __restrict__ vT,
                     _Float16* __restrict__ ao) {
  __shared__ __align__(16) _Float16 Ps[8][16][72];  // per-wave private
  const int h  = blockIdx.x;
  const int qb = blockIdx.y;
  const int q0 = qb * 128;
  const int tid = threadIdx.x;
  const int lane = tid & 31, wid = tid >> 5;
  const int lg = lane >> 4, ll = lane & 15;
  const int qrow = q0 + wid * 16 + ll;   // A-frag row

  // Q fragments: 6 k-steps over DK=192
  v16h qa[6];
  const _Float16* qp = qf + ((size_t)qrow * H_N + h) * DK_N;
  for (int ks = 0; ks < 6; ++ks) {
    uint4* d = reinterpret_cast<uint4*>(&qa[ks]);
    d[0] = *reinterpret_cast<const uint4*>(qp + ks * 32 + 16 * lg);
    d[1] = *reinterpret_cast<const uint4*>(qp + ks * 32 + 8 + 16 * lg);
  }

  v8f o[8];
  for (int nf = 0; nf < 8; ++nf)
    for (int i = 0; i < 8; ++i) o[nf][i] = 0.0f;
  float mrow[8], lrow[8];
  for (int v = 0; v < 8; ++v) { mrow[v] = -3.0e38f; lrow[v] = 0.0f; }

  for (int kb = 0; kb <= 2 * qb + 1; ++kb) {
    const int kbase = kb * 64;
    // S = Q * K^T  (16 x 64 per wave)
    v8f sfrag[4];
    for (int nf = 0; nf < 4; ++nf) {
      v8f s;
      for (int i = 0; i < 8; ++i) s[i] = 0.0f;
      int key = kbase + nf * 16 + ll;
      const _Float16* kp = kf + ((size_t)key * H_N + h) * DK_N + 16 * lg;
      for (int ks = 0; ks < 6; ++ks) {
        v16h kb_frag;
        uint4* d = reinterpret_cast<uint4*>(&kb_frag);
        d[0] = *reinterpret_cast<const uint4*>(kp + ks * 32);
        d[1] = *reinterpret_cast<const uint4*>(kp + ks * 32 + 8);
        s = __builtin_amdgcn_wmma_f32_16x16x32_f16(false, qa[ks], false,
                                                   kb_frag, (short)0, s,
                                                   false, false);
      }
      sfrag[nf] = s;
    }
    // causal mask (only diagonal-ish key blocks)
    if (kbase + 64 > q0) {
      for (int nf = 0; nf < 4; ++nf) {
        int key = kbase + nf * 16 + ll;
        for (int v = 0; v < 8; ++v) {
          int row = q0 + wid * 16 + 8 * lg + v;
          if (key > row) sfrag[nf][v] = -3.0e38f;
        }
      }
    }
    // online softmax (reduce across the 16 lanes of each half-wave)
    for (int v = 0; v < 8; ++v) {
      float m = sfrag[0][v];
      for (int nf = 1; nf < 4; ++nf) m = fmaxf(m, sfrag[nf][v]);
      for (int msk = 1; msk < 16; msk <<= 1) m = fmaxf(m, __shfl_xor(m, msk, 32));
      float mnew = fmaxf(mrow[v], m);
      float sc = __expf(mrow[v] - mnew);
      mrow[v] = mnew;
      float rs = 0.0f;
      for (int nf = 0; nf < 4; ++nf) {
        float p = __expf(sfrag[nf][v] - mnew);
        sfrag[nf][v] = p;
        rs += p;
      }
      for (int msk = 1; msk < 16; msk <<= 1) rs += __shfl_xor(rs, msk, 32);
      lrow[v] = lrow[v] * sc + rs;
      for (int nf = 0; nf < 8; ++nf) o[nf][v] *= sc;
    }
    // P (C-layout) -> LDS -> A-layout; per-wave private, in-order LDS
    for (int nf = 0; nf < 4; ++nf)
      for (int v = 0; v < 8; ++v)
        Ps[wid][8 * lg + v][nf * 16 + ll] = (_Float16)sfrag[nf][v];
    v16h pa[2];
    for (int ks2 = 0; ks2 < 2; ++ks2) {
      uint4* d = reinterpret_cast<uint4*>(&pa[ks2]);
      d[0] = *reinterpret_cast<const uint4*>(&Ps[wid][ll][ks2 * 32 + 16 * lg]);
      d[1] = *reinterpret_cast<const uint4*>(&Ps[wid][ll][ks2 * 32 + 8 + 16 * lg]);
    }
    // O += P * V
    for (int nf = 0; nf < 8; ++nf) {
      int dv = nf * 16 + ll;
      const _Float16* vp = vT + ((size_t)h * DV_N + dv) * S_LEN + kbase + 16 * lg;
      for (int ks2 = 0; ks2 < 2; ++ks2) {
        v16h vb;
        uint4* d = reinterpret_cast<uint4*>(&vb);
        d[0] = *reinterpret_cast<const uint4*>(vp + ks2 * 32);
        d[1] = *reinterpret_cast<const uint4*>(vp + ks2 * 32 + 8);
        o[nf] = __builtin_amdgcn_wmma_f32_16x16x32_f16(false, pa[ks2], false,
                                                       vb, (short)0, o[nf],
                                                       false, false);
      }
    }
  }

  // write ao [S][H*DV] as f16 (input to the o-proj GEMM)
  for (int nf = 0; nf < 8; ++nf) {
    int dv = nf * 16 + ll;
    for (int v = 0; v < 8; ++v) {
      int row = q0 + wid * 16 + 8 * lg + v;
      ao[(size_t)row * HD_COLS + h * DV_N + dv] = (_Float16)(o[nf][v] / lrow[v]);
    }
  }
}

// ---------------------------------------------------------------------------
// Router: logits = hm @ router_w, softmax, top-2, normalized weights.
// top-2 of softmax == top-2 of logits; weights from top-2 logits directly.
// ---------------------------------------------------------------------------
__global__ __launch_bounds__(256)
void router_topk_kernel(const _Float16* __restrict__ hm,
                        const float* __restrict__ rw,
                        float* __restrict__ combine) {
  int t = blockIdx.x;
  float acc[E_N];
  for (int e = 0; e < E_N; ++e) acc[e] = 0.0f;
  for (int d = threadIdx.x; d < D_DIM; d += 256) {
    float hv = (float)hm[(size_t)t * D_DIM + d];
    const float* r = rw + (size_t)d * E_N;
    for (int e = 0; e < E_N; ++e) acc[e] += hv * r[e];
  }
  __shared__ float red[8][E_N];
  for (int e = 0; e < E_N; ++e)
    for (int m = 16; m >= 1; m >>= 1) acc[e] += __shfl_xor(acc[e], m, 32);
  if ((threadIdx.x & 31) == 0)
    for (int e = 0; e < E_N; ++e) red[threadIdx.x >> 5][e] = acc[e];
  __syncthreads();
  if (threadIdx.x == 0) {
    float lg[E_N];
    for (int e = 0; e < E_N; ++e) {
      float s = 0.0f;
      for (int w = 0; w < 8; ++w) s += red[w][e];
      lg[e] = s;
    }
    int i1 = 0;
    for (int e = 1; e < E_N; ++e) if (lg[e] > lg[i1]) i1 = e;
    int i2 = (i1 == 0) ? 1 : 0;
    for (int e = 0; e < E_N; ++e) if (e != i1 && lg[e] > lg[i2]) i2 = e;
    float e2 = __expf(lg[i2] - lg[i1]);
    float w1 = 1.0f / (1.0f + e2);
    float w2 = e2 / (1.0f + e2);
    for (int e = 0; e < E_N; ++e) combine[(size_t)t * E_N + e] = 0.0f;
    combine[(size_t)t * E_N + i1] = w1;
    combine[(size_t)t * E_N + i2] = w2;
  }
}

// ---------------------------------------------------------------------------
// Host orchestration
// ---------------------------------------------------------------------------
extern "C" void kernel_launch(void* const* d_in, const int* in_sizes, int n_in,
                              void* d_out, int out_size, void* d_ws,
                              size_t ws_size, hipStream_t stream) {
  (void)in_sizes; (void)n_in; (void)out_size; (void)ws_size;
  const float* x           = (const float*)d_in[0];
  const float* norm1_w     = (const float*)d_in[1];
  const float* norm2_w     = (const float*)d_in[2];
  const float* q_a_w       = (const float*)d_in[3];
  const float* q_a_norm_w  = (const float*)d_in[4];
  const float* q_b_w       = (const float*)d_in[5];
  const float* kv_a_w      = (const float*)d_in[6];
  const float* kv_a_norm_w = (const float*)d_in[7];
  const float* kv_b_w      = (const float*)d_in[8];
  const float* o_w         = (const float*)d_in[9];
  const float* router_w    = (const float*)d_in[10];
  const float* w_gate      = (const float*)d_in[11];
  const float* w_up        = (const float*)d_in[12];
  const float* w_down      = (const float*)d_in[13];
  float* out = (float*)d_out;

  char* ws = (char*)d_ws;
  size_t off = 0;
  auto alloc = [&](size_t bytes) -> void* {
    off = (off + 255) & ~(size_t)255;
    void* p = ws + off;
    off += bytes;
    return p;
  };

  const size_t SD = (size_t)S_LEN * D_DIM;
  _Float16* hb16     = (_Float16*)alloc(SD * 2);
  _Float16* w_qa16   = (_Float16*)alloc((size_t)D_DIM * QR_N * 2);
  _Float16* w_qb16   = (_Float16*)alloc((size_t)QR_N * QB_COLS * 2);
  _Float16* w_kva16  = (_Float16*)alloc((size_t)D_DIM * KVA_COLS * 2);
  _Float16* w_kvb16  = (_Float16*)alloc((size_t)KVR_N * KVB_COLS * 2);
  _Float16* w_o16    = (_Float16*)alloc((size_t)HD_COLS * D_DIM * 2);
  _Float16* w_gate16 = (_Float16*)alloc((size_t)E_N * D_DIM * F_N * 2);
  _Float16* w_up16   = (_Float16*)alloc((size_t)E_N * D_DIM * F_N * 2);
  _Float16* w_down16 = (_Float16*)alloc((size_t)E_N * F_N * D_DIM * 2);
  float*    qa_f32   = (float*)alloc((size_t)S_LEN * QR_N * 4);
  _Float16* qa16n    = (_Float16*)alloc((size_t)S_LEN * QR_N * 2);
  float*    q_f32    = (float*)alloc((size_t)S_LEN * QB_COLS * 4);
  float*    ckv_f32  = (float*)alloc((size_t)S_LEN * KVA_COLS * 4);
  _Float16* c16n     = (_Float16*)alloc((size_t)S_LEN * KVR_N * 2);
  float*    kv_f32   = (float*)alloc((size_t)S_LEN * KVB_COLS * 4);
  _Float16* qf16     = (_Float16*)alloc((size_t)S_LEN * H_N * DK_N * 2);
  _Float16* kf16     = (_Float16*)alloc((size_t)S_LEN * H_N * DK_N * 2);
  _Float16* vT16     = (_Float16*)alloc((size_t)H_N * DV_N * S_LEN * 2);
  _Float16* ao16     = (_Float16*)alloc(SD * 2);
  float*    attn_f32 = (float*)alloc(SD * 4);
  float*    x1_f32   = (float*)alloc(SD * 4);
  _Float16* hm16     = (_Float16*)alloc(SD * 2);
  float*    combine  = (float*)alloc((size_t)S_LEN * E_N * 4);
  float*    g_f32    = (float*)alloc((size_t)S_LEN * F_N * 4);
  float*    u_f32    = (float*)alloc((size_t)S_LEN * F_N * 4);
  _Float16* a16      = (_Float16*)alloc((size_t)S_LEN * F_N * 2);
  float*    eo_f32   = (float*)alloc(SD * 4);
  float*    mo_f32   = (float*)alloc(SD * 4);

  auto conv = [&](const float* src, _Float16* dst, size_t n) {
    f32_to_f16_kernel<<<2048, 256, 0, stream>>>(src, dst, n);
  };
  auto gemm = [&](const _Float16* A, const _Float16* B, float* C, int M, int N,
                  int K) {
    dim3 grid((N + 127) / 128, M / 128);
    gemm_f16_kernel<<<grid, 256, 0, stream>>>(A, B, C, M, N, K);
  };

  // 0) weight conversion fp32 -> f16 (done every call; deterministic)
  conv(q_a_w, w_qa16, (size_t)D_DIM * QR_N);
  conv(q_b_w, w_qb16, (size_t)QR_N * QB_COLS);
  conv(kv_a_w, w_kva16, (size_t)D_DIM * KVA_COLS);
  conv(kv_b_w, w_kvb16, (size_t)KVR_N * KVB_COLS);
  conv(o_w, w_o16, (size_t)HD_COLS * D_DIM);
  conv(w_gate, w_gate16, (size_t)E_N * D_DIM * F_N);
  conv(w_up, w_up16, (size_t)E_N * D_DIM * F_N);
  conv(w_down, w_down16, (size_t)E_N * F_N * D_DIM);

  // 1) h = rmsnorm(x, norm1_w) -> f16
  rmsnorm_f16_kernel<<<S_LEN, 256, 0, stream>>>(x, D_DIM, norm1_w, hb16, D_DIM, D_DIM);

  // 2) q path
  gemm(hb16, w_qa16, qa_f32, S_LEN, QR_N, D_DIM);
  rmsnorm_f16_kernel<<<S_LEN, 256, 0, stream>>>(qa_f32, QR_N, q_a_norm_w, qa16n, QR_N, QR_N);
  gemm(qa16n, w_qb16, q_f32, S_LEN, QB_COLS, QR_N);

  // 3) kv path
  gemm(hb16, w_kva16, ckv_f32, S_LEN, KVA_COLS, D_DIM);
  rmsnorm_f16_kernel<<<S_LEN, 256, 0, stream>>>(ckv_f32, KVA_COLS, kv_a_norm_w, c16n, KVR_N, KVR_N);
  gemm(c16n, w_kvb16, kv_f32, S_LEN, KVB_COLS, KVR_N);

  // 4) RoPE + layouts
  prep_qkv_kernel<<<S_LEN, 256, 0, stream>>>(q_f32, kv_f32, ckv_f32, qf16, kf16, vT16);

  // 5) flash attention
  {
    dim3 grid(H_N, S_LEN / 128);
    mla_attn_kernel<<<grid, 256, 0, stream>>>(qf16, kf16, vT16, ao16);
  }

  // 6) o-proj + residual
  gemm(ao16, w_o16, attn_f32, S_LEN, D_DIM, HD_COLS);
  add_f32_kernel<<<2048, 256, 0, stream>>>(x, attn_f32, x1_f32, SD);

  // 7) norm2 + router
  rmsnorm_f16_kernel<<<S_LEN, 256, 0, stream>>>(x1_f32, D_DIM, norm2_w, hm16, D_DIM, D_DIM);
  router_topk_kernel<<<S_LEN, 256, 0, stream>>>(hm16, router_w, combine);

  // 8) MoE (dense over experts, as in reference), accumulate weighted
  hipMemsetAsync(mo_f32, 0, SD * 4, stream);
  for (int e = 0; e < E_N; ++e) {
    const _Float16* wg = w_gate16 + (size_t)e * D_DIM * F_N;
    const _Float16* wu = w_up16 + (size_t)e * D_DIM * F_N;
    const _Float16* wd = w_down16 + (size_t)e * F_N * D_DIM;
    gemm(hm16, wg, g_f32, S_LEN, F_N, D_DIM);
    gemm(hm16, wu, u_f32, S_LEN, F_N, D_DIM);
    silu_mul_kernel<<<2048, 256, 0, stream>>>(g_f32, u_f32, a16, (size_t)S_LEN * F_N);
    gemm(a16, wd, eo_f32, S_LEN, D_DIM, F_N);
    moe_accum_kernel<<<2048, 256, 0, stream>>>(eo_f32, combine, mo_f32, e, SD);
  }

  // 9) final residual
  add_f32_kernel<<<2048, 256, 0, stream>>>(x1_f32, mo_f32, out, SD);
}
